// GCN_44263932952673
// MI455X (gfx1250) — matrix operand order repaired and checked by reading, compiled, not run on hardware
//
#include <hip/hip_runtime.h>

// GCN forward, MI455X (gfx1250, wave32).
// out = colsum(adj) @ relu(adj @ (x@W1) + b1) @ W2 + N*b2, then @ W_out + b_out.
// Single streamed pass over the 1 GiB adj matrix; f16 WMMA, f32 accumulate,
// software-pipelined loads so WMMA never drains loadcnt to zero.

#define NN   16384
#define F_IN 128
#define F_HID 128
#define F_OUT 64

typedef __attribute__((ext_vector_type(16))) _Float16 v16h;
typedef __attribute__((ext_vector_type(8)))  _Float16 v8h;
typedef __attribute__((ext_vector_type(8)))  float    v8f;

__device__ __forceinline__ v8f wmma_f16(v16h a, v16h b, v8f c) {
    // D = A(16x32) * B(32x16) + C, f32 accumulate
    return __builtin_amdgcn_wmma_f32_16x16x32_f16(
        /*neg_a=*/false, a, /*neg_b=*/false, b,
        /*c_mod=*/(short)0, c, /*reuse_a=*/false, /*reuse_b=*/false);
}

// A-tile (16x32) load: f32 row-major source, convert to f16 WMMA layout.
// lane half h, lane idx: elements e<8 -> K=e+8h ; e>=8 -> K=e+8+8h.
__device__ __forceinline__ v16h load_a_f32(const float* ap) {
    float4 fa = *(const float4*)(ap);
    float4 fb = *(const float4*)(ap + 4);
    float4 fc = *(const float4*)(ap + 16);
    float4 fd = *(const float4*)(ap + 20);
    float av[16] = {fa.x, fa.y, fa.z, fa.w, fb.x, fb.y, fb.z, fb.w,
                    fc.x, fc.y, fc.z, fc.w, fd.x, fd.y, fd.z, fd.w};
    v16h a;
#pragma unroll
    for (int e = 0; e < 16; ++e) a[e] = (_Float16)av[e];
    return a;
}

// B-tile (32x16) load from f16 "transposed" matrix: e<8 -> K=e+8h ; e>=8 -> K=e+8+8h.
__device__ __forceinline__ v16h load_b_f16(const _Float16* bp) {
    v8h lo = *(const v8h*)(bp);
    v8h hi = *(const v8h*)(bp + 16);
    v16h b;
#pragma unroll
    for (int e = 0; e < 8; ++e) { b[e] = lo[e]; b[e + 8] = hi[e]; }
    return b;
}

// ---------------------------------------------------------------- kernel 0
// W1T[f][k] = (f16) W1[k][f]   (B-operand friendly layout for kernel 1)
__global__ void k0_w1t(const float* __restrict__ W1, _Float16* __restrict__ W1T) {
    int i = blockIdx.x * 256 + threadIdx.x;      // 16384 elems
    int f = i >> 7, k = i & 127;
    W1T[f * F_IN + k] = (_Float16)W1[k * F_HID + f];
}

// ---------------------------------------------------------------- kernel 1
// supportT[f][node] = (f16)( x @ W1 )[node][f]    via WMMA, K=128 (4 steps)
__global__ __launch_bounds__(256) void k1_xw(const float* __restrict__ x,
                                             const _Float16* __restrict__ W1T,
                                             _Float16* __restrict__ sT) {
    const int wv   = threadIdx.x >> 5;
    const int lane = threadIdx.x & 31;
    const int h    = lane >> 4;         // lane half
    const int idx  = lane & 15;
    const int r0   = blockIdx.x * 128 + wv * 16;   // 16 node-rows per wave

    const float* arow = x + (size_t)(r0 + idx) * F_IN;
    v8f acc[8] = {};                    // 16 rows x 128 feats = 8 C tiles

#pragma unroll
    for (int k0 = 0; k0 < F_IN; k0 += 32) {
        v16h a = load_a_f32(arow + k0 + 8 * h);
        v16h b[8];
#pragma unroll
        for (int t = 0; t < 8; ++t)
            b[t] = load_b_f16(W1T + (size_t)(16 * t + idx) * F_IN + k0 + 8 * h);
#pragma unroll
        for (int t = 0; t < 8; ++t)
            acc[t] = wmma_f16(a, b[t], acc[t]);
    }
    // store transposed: supportT[feat][node]
#pragma unroll
    for (int t = 0; t < 8; ++t) {
#pragma unroll
        for (int r = 0; r < 8; ++r) {
            int node = r0 + r + 8 * h;          // C: M = r + 8*half
            int feat = 16 * t + idx;            // C: N = idx
            sT[(size_t)feat * NN + node] = (_Float16)acc[t][r];
        }
    }
}

// ---------------------------------------------------------------- kernel 2
// The 1 GiB streaming pass (software pipelined):
//   h1[node][f] = relu( (adj @ support1)[node][f] + b1[f] )      (f16 out)
//   cpart[blk][k] = sum over this block's 64 rows of adj[., k]   (colsum partial)
__global__ __launch_bounds__(256) void k2_layer1(const float* __restrict__ adj,
                                                 const _Float16* __restrict__ sT,
                                                 const float* __restrict__ b1,
                                                 _Float16* __restrict__ h1,
                                                 float* __restrict__ cpart) {
    __shared__ float lds_c[NN];                 // 64 KB per-block colsum partial
    const int tid = threadIdx.x;
    for (int i = tid; i < NN; i += 256) lds_c[i] = 0.f;
    __syncthreads();

    const int wv   = tid >> 5;
    const int lane = tid & 31;
    const int mw   = wv >> 1;                   // 4 row groups of 16
    const int nh   = wv & 1;                    // 2 feature halves of 64
    const int h    = lane >> 4;
    const int idx  = lane & 15;
    const int r0   = blockIdx.x * 64 + mw * 16;
    const int f0   = nh * 64;

    const float*    arow = adj + (size_t)(r0 + idx) * NN + 8 * h;
    const _Float16* brow = sT + (size_t)(f0 + idx) * NN + 8 * h;
    v8f acc[4] = {};

    // ---- pipeline prologue: stage 0 in flight
    v16h aCur = load_a_f32(arow);
    v16h bCur[4], aNxt = {}, bNxt[4] = {};
#pragma unroll
    for (int t = 0; t < 4; ++t) bCur[t] = load_b_f16(brow + (size_t)(16 * t) * NN);

#pragma unroll 2
    for (int k0 = 0; k0 < NN; k0 += 32) {
        const int k1 = k0 + 32;
        // ---- issue next stage's loads before consuming current stage
        if (k1 < NN) {
            __builtin_prefetch(arow + k1 + 256, 0, 1);   // global_prefetch ~1KB ahead
            aNxt = load_a_f32(arow + k1);
#pragma unroll
            for (int t = 0; t < 4; ++t)
                bNxt[t] = load_b_f16(brow + (size_t)(16 * t) * NN + k1);
        }

        // ---- 4 WMMAs on current stage (operands already resident)
#pragma unroll
        for (int t = 0; t < 4; ++t)
            acc[t] = wmma_f16(aCur, bCur[t], acc[t]);

        // ---- colsum(adj) contribution from the resident A tile.
        // A is loaded identically by nh=0 and nh=1 waves; alternate duty.
        if (((k0 >> 5) & 1) == nh) {
            float mysum = 0.f;
#pragma unroll
            for (int e = 0; e < 16; ++e) {
                float s = (float)aCur[e];        // 16 rows of column col(e,h) live
                s += __shfl_xor(s, 1, 32);       //   in lanes of this 16-lane half
                s += __shfl_xor(s, 2, 32);
                s += __shfl_xor(s, 4, 32);
                s += __shfl_xor(s, 8, 32);
                if (idx == e) mysum = s;         // lane idx owns column of elem idx
            }
            int col = k0 + (idx < 8 ? idx + 8 * h : idx + 8 + 8 * h); // bijective 0..31
            atomicAdd(&lds_c[col], mysum);       // ds_add_f32
        }

        // ---- rotate pipeline (pure renaming under unroll-2)
        aCur = aNxt;
#pragma unroll
        for (int t = 0; t < 4; ++t) bCur[t] = bNxt[t];
    }

    // ---- epilogue: bias + relu, store h1 row-major f16
#pragma unroll
    for (int t = 0; t < 4; ++t) {
#pragma unroll
        for (int r = 0; r < 8; ++r) {
            int node = r0 + r + 8 * h;
            int feat = f0 + 16 * t + idx;
            float v = acc[t][r] + b1[feat];
            v = fmaxf(v, 0.f);
            h1[(size_t)node * F_HID + feat] = (_Float16)v;
        }
    }

    __syncthreads();
    float* cp = cpart + (size_t)blockIdx.x * NN;
    for (int i = tid; i < NN; i += 256) cp[i] = lds_c[i];
}

// ---------------------------------------------------------------- kernel 3a
// c[k] = sum over 256 block partials (deterministic fixed order)
__global__ void k3a_csum(const float* __restrict__ cpart, float* __restrict__ c) {
    int k = blockIdx.x * 256 + threadIdx.x;
    float s = 0.f;
    for (int b = 0; b < 256; ++b) s += cpart[(size_t)b * NN + k];
    c[k] = s;
}

// ---------------------------------------------------------------- kernel 3b
// qpart[blk][f] = sum_{k in chunk} c[k] * h1[k][f]
__global__ void k3b_q(const float* __restrict__ c, const _Float16* __restrict__ h1,
                      float* __restrict__ qpart) {
    int f  = threadIdx.x;                        // 128 feats
    int kb = blockIdx.x;                         // 128 chunks of 128 nodes
    float s = 0.f;
    for (int k = kb * 128; k < kb * 128 + 128; ++k)
        s += c[k] * (float)h1[(size_t)k * F_HID + f];
    qpart[kb * 128 + f] = s;
}

// ---------------------------------------------------------------- kernel 3c
// q = reduce(qpart); pooled = q@W2 + N*b2; out = pooled@W_out + b_out
__global__ void k3c_final(const float* __restrict__ qpart, const float* __restrict__ W2,
                          const float* __restrict__ b2, const float* __restrict__ Wout,
                          const float* __restrict__ bout, float* __restrict__ out) {
    __shared__ float qf[F_HID];
    __shared__ float red[F_OUT];
    int t = threadIdx.x;                         // 128 threads
    float s = 0.f;
    for (int b = 0; b < 128; ++b) s += qpart[b * 128 + t];
    qf[t] = s;
    __syncthreads();
    if (t < F_OUT) {
        float p = (float)NN * b2[t];
        for (int f = 0; f < F_HID; ++f) p += qf[f] * W2[f * F_OUT + t];
        red[t] = p * Wout[t];
    }
    __syncthreads();
    if (t == 0) {
        float o = bout[0];
        for (int n = 0; n < F_OUT; ++n) o += red[n];
        out[0] = o;
    }
}

// ---------------------------------------------------------------- launch
extern "C" void kernel_launch(void* const* d_in, const int* in_sizes, int n_in,
                              void* d_out, int out_size, void* d_ws, size_t ws_size,
                              hipStream_t stream) {
    (void)in_sizes; (void)n_in; (void)out_size; (void)ws_size;
    const float* x    = (const float*)d_in[0];
    const float* adj  = (const float*)d_in[1];
    const float* W1   = (const float*)d_in[2];
    const float* b1   = (const float*)d_in[3];
    const float* W2   = (const float*)d_in[4];
    const float* b2   = (const float*)d_in[5];
    const float* Wout = (const float*)d_in[6];
    const float* bout = (const float*)d_in[7];
    float* out = (float*)d_out;

    // workspace layout (~24.2 MB total)
    char* ws = (char*)d_ws;
    _Float16* W1T   = (_Float16*)(ws);                                    // 32 KB
    _Float16* sT    = (_Float16*)(ws + (32u << 10));                      // 4 MB
    _Float16* h1    = (_Float16*)(ws + (32u << 10) + (4u << 20));         // 4 MB
    float*    cpart = (float*)   (ws + (32u << 10) + (8u << 20));         // 16 MB
    float*    c     = (float*)   (ws + (32u << 10) + (24u << 20));        // 64 KB
    float*    qpart = (float*)   (ws + (32u << 10) + (24u << 20) + (64u << 10)); // 64 KB

    k0_w1t   <<<  64, 256, 0, stream>>>(W1, W1T);
    k1_xw    <<< 128, 256, 0, stream>>>(x, W1T, sT);
    k2_layer1<<< 256, 256, 0, stream>>>(adj, sT, b1, h1, cpart);
    k3a_csum <<<  64, 256, 0, stream>>>(cpart, c);
    k3b_q    <<< 128, 128, 0, stream>>>(c, h1, qpart);
    k3c_final<<<   1, 128, 0, stream>>>(qpart, W2, b2, Wout, bout, out);
}
